// ATKT_20014547599415
// MI455X (gfx1250) — matrix operand order repaired and checked by reading, not compiled, not run on hardware
//
#include <hip/hip_runtime.h>
#include <hip/hip_bf16.h>

typedef __attribute__((ext_vector_type(16))) _Float16 v16h;
typedef __attribute__((ext_vector_type(8)))  float    v8f;

#define B_   64
#define T_   512
#define NUMC 2000
#define SK_  256
#define AN_  96
#define H_   256
#define ATT_ 80
#define D_   352              // SK + AN
#define M_   (B_ * T_)        // 32768 rows
#define G4H  (4 * H_)         // 1024

__device__ __forceinline__ float sigmoidf_(float x) {
    return 1.0f / (1.0f + __expf(-x));
}

// ---------------------------------------------------------------- converts
__global__ void k_cvt_f16(const float* __restrict__ src, _Float16* __restrict__ dst, int n) {
    for (int i = blockIdx.x * blockDim.x + threadIdx.x; i < n; i += gridDim.x * blockDim.x)
        dst[i] = (_Float16)src[i];
}

__global__ void k_bias_sum(const float* __restrict__ bih, const float* __restrict__ bhh,
                           float* __restrict__ out) {
    int i = blockIdx.x * blockDim.x + threadIdx.x;
    if (i < G4H) out[i] = bih[i] + bhh[i];
}

// ---------------------------------------------------------------- embeddings
// sae[m,d] = answer==1 ? [se|ae] : [ae|se]; write f32 to d_out tail + f16 for GEMM
__global__ void k_embed(const int* __restrict__ skill, const int* __restrict__ answer,
                        const float* __restrict__ skill_emb, const float* __restrict__ answer_emb,
                        float* __restrict__ sae_out, _Float16* __restrict__ sae_h) {
    int m   = blockIdx.x;
    int ans = answer[m];
    int sk  = skill[m];
    const float* se = skill_emb  + (size_t)sk  * SK_;
    const float* ae = answer_emb + (size_t)ans * AN_;
    for (int d = threadIdx.x; d < D_; d += blockDim.x) {
        float v;
        if (ans == 1) v = (d < SK_) ? se[d] : ae[d - SK_];
        else          v = (d < AN_) ? ae[d] : se[d - AN_];
        sae_out[(size_t)m * D_ + d] = v;
        sae_h [(size_t)m * D_ + d] = (_Float16)v;
    }
}

// ---------------------------------------------------------------- xg = sae @ Wih^T + (bih+bhh)
// M=32768, N=1024, K=352 (11 k-steps of 32). Block: 8 waves, A tile staged in LDS.
__global__ __launch_bounds__(256) void k_xg(const _Float16* __restrict__ saeh,
                                            const _Float16* __restrict__ Wihh,
                                            const float* __restrict__ biasSum,
                                            float* __restrict__ xg) {
    __shared__ _Float16 lds_a[16 * D_];
    int mtile = blockIdx.x;
    for (int i = threadIdx.x; i < 16 * D_; i += 256) {
        int r = i / D_, k = i - r * D_;
        lds_a[i] = saeh[((size_t)(mtile * 16 + r)) * D_ + k];
    }
    __syncthreads();

    int wave = threadIdx.x >> 5;
    int lane = threadIdx.x & 31;
    int ln = lane & 15, half = lane >> 4;
    int ntile = blockIdx.y * 8 + wave;
    const _Float16* bsrc = Wihh + ((size_t)(ntile * 16 + ln)) * D_;  // B(k,n)=Wih[n][k]

    v8f c = {};
    for (int ks = 0; ks < 11; ++ks) {
        int kb = ks * 32;
        v16h a, b;
        #pragma unroll
        for (int e = 0; e < 16; ++e) {
            a[e] = lds_a[ln * D_ + kb + (e & 7) + 8 * half + 16 * (e >> 3)];
            b[e] = bsrc[kb + (e & 7) + 8 * half + 16 * (e >> 3)];
        }
        c = __builtin_amdgcn_wmma_f32_16x16x32_f16(false, a, false, b, (short)0, c, false, false);
    }
    int col = ntile * 16 + ln;
    float bs = biasSum[col];
    #pragma unroll
    for (int v = 0; v < 8; ++v) {
        int row = mtile * 16 + v + 8 * half;
        xg[(size_t)row * G4H + col] = c[v] + bs;
    }
}

// ---------------------------------------------------------------- LSTM recurrence
// grid = 4 blocks (16 batch rows each), block = 512 threads = 16 waves.
// Wave w owns hidden tile w (cols w*16..w*16+15) for ALL 4 gates; c-state in regs.
__global__ __launch_bounds__(512) void k_lstm(const float* __restrict__ xg,
                                              const _Float16* __restrict__ Whhh,
                                              float* __restrict__ h_all,
                                              _Float16* __restrict__ h_allh) {
    __shared__ _Float16 lds_h[16 * H_];   // h_{t-1} in f16, rows=batch-in-block
    int wave = threadIdx.x >> 5;
    int lane = threadIdx.x & 31;
    int ln = lane & 15, half = lane >> 4;
    int bblk = blockIdx.x;

    for (int i = threadIdx.x; i < 16 * H_; i += 512) lds_h[i] = (_Float16)0.0f;
    __syncthreads();

    float cst[8];
    #pragma unroll
    for (int v = 0; v < 8; ++v) cst[v] = 0.0f;

    const _Float16* bp0 = Whhh + ((size_t)(0 * H_ + wave * 16 + ln)) * H_;
    const _Float16* bp1 = Whhh + ((size_t)(1 * H_ + wave * 16 + ln)) * H_;
    const _Float16* bp2 = Whhh + ((size_t)(2 * H_ + wave * 16 + ln)) * H_;
    const _Float16* bp3 = Whhh + ((size_t)(3 * H_ + wave * 16 + ln)) * H_;

    int hid = wave * 16 + ln;

    for (int t = 0; t < T_; ++t) {
        v8f a0 = {}, a1 = {}, a2 = {}, a3 = {};
        for (int ks = 0; ks < 8; ++ks) {
            int kb = ks * 32;
            v16h a;
            #pragma unroll
            for (int e = 0; e < 16; ++e)
                a[e] = lds_h[ln * H_ + kb + (e & 7) + 8 * half + 16 * (e >> 3)];
            v16h b0, b1, b2, b3;
            #pragma unroll
            for (int e = 0; e < 16; ++e) {
                int ko = kb + (e & 7) + 8 * half + 16 * (e >> 3);
                b0[e] = bp0[ko]; b1[e] = bp1[ko]; b2[e] = bp2[ko]; b3[e] = bp3[ko];
            }
            a0 = __builtin_amdgcn_wmma_f32_16x16x32_f16(false, a, false, b0, (short)0, a0, false, false);
            a1 = __builtin_amdgcn_wmma_f32_16x16x32_f16(false, a, false, b1, (short)0, a1, false, false);
            a2 = __builtin_amdgcn_wmma_f32_16x16x32_f16(false, a, false, b2, (short)0, a2, false, false);
            a3 = __builtin_amdgcn_wmma_f32_16x16x32_f16(false, a, false, b3, (short)0, a3, false, false);
        }
        float hnv[8];
        #pragma unroll
        for (int v = 0; v < 8; ++v) {
            int m = v + 8 * half;
            int batch = bblk * 16 + m;
            size_t rowi  = (size_t)batch * T_ + t;
            size_t gbase = rowi * G4H;
            float gi = a0[v] + xg[gbase + 0 * H_ + hid];
            float gf = a1[v] + xg[gbase + 1 * H_ + hid];
            float gg = a2[v] + xg[gbase + 2 * H_ + hid];
            float go = a3[v] + xg[gbase + 3 * H_ + hid];
            float cn = sigmoidf_(gf) * cst[v] + sigmoidf_(gi) * tanhf(gg);
            cst[v] = cn;
            float hn = sigmoidf_(go) * tanhf(cn);
            hnv[v] = hn;
            h_all [rowi * H_ + hid] = hn;
            h_allh[rowi * H_ + hid] = (_Float16)hn;
        }
        __syncthreads();   // all waves done reading lds_h for step t
        #pragma unroll
        for (int v = 0; v < 8; ++v)
            lds_h[(v + 8 * half) * H_ + hid] = (_Float16)hnv[v];
        __syncthreads();
    }
}

// ---------------------------------------------------------------- attention scores
__global__ __launch_bounds__(128) void k_scores(const float* __restrict__ h_all,
                                                const float* __restrict__ mlp_w,
                                                const float* __restrict__ mlp_b,
                                                const float* __restrict__ sim_w,
                                                float* __restrict__ scores) {
    __shared__ float red[128];
    int m = blockIdx.x;
    int a = threadIdx.x;
    float val = 0.0f;
    if (a < ATT_) {
        const float* hr = h_all + (size_t)m * H_;
        const float* wr = mlp_w + (size_t)a * H_;
        float dot = mlp_b[a];
        for (int k = 0; k < H_; ++k) dot += hr[k] * wr[k];
        val = tanhf(dot) * sim_w[a];
    }
    red[a] = val;
    __syncthreads();
    for (int s = 64; s > 0; s >>= 1) {
        if (a < s) red[a] += red[a + s];
        __syncthreads();
    }
    if (a == 0) scores[m] = red[0];
}

// ---------------------------------------------------------------- causal scan per batch row
// attn = cumsum(e*h)/cumsum(e);  cum1 = exclusive cumsum(attn) -> f16 for FC
__global__ __launch_bounds__(256) void k_rowscan(const float* __restrict__ scores,
                                                 const float* __restrict__ h_all,
                                                 _Float16* __restrict__ cum1h) {
    __shared__ float ev[T_];
    __shared__ float den[T_];
    __shared__ float red[256];
    int b = blockIdx.x;
    const float* sc = scores + (size_t)b * T_;

    float mx = -1e30f;
    for (int t = threadIdx.x; t < T_; t += 256) mx = fmaxf(mx, sc[t]);
    red[threadIdx.x] = mx;
    __syncthreads();
    for (int s = 128; s > 0; s >>= 1) {
        if (threadIdx.x < s) red[threadIdx.x] = fmaxf(red[threadIdx.x], red[threadIdx.x + s]);
        __syncthreads();
    }
    mx = red[0];
    for (int t = threadIdx.x; t < T_; t += 256) ev[t] = __expf(sc[t] - mx);
    __syncthreads();
    if (threadIdx.x == 0) {
        float s = 0.0f;
        for (int t = 0; t < T_; ++t) { s += ev[t]; den[t] = s; }
    }
    __syncthreads();

    int j = threadIdx.x;  // hidden dim, 256 == H
    float accn = 0.0f, c1 = 0.0f;
    for (int t = 0; t < T_; ++t) {
        size_t idx = ((size_t)b * T_ + t) * H_ + j;
        float ht = h_all[idx];
        accn += ev[t] * ht;
        float attn = accn / den[t];
        cum1h[idx] = (_Float16)c1;   // exclusive cumsum
        c1 += attn;
    }
}

// ---------------------------------------------------------------- res = sigmoid([cum1|h] @ fc_w^T + fc_b)
// M=32768, N=2000 (125 tiles), K=512 (16 k-steps).
// B tile (16x512) staged in LDS once per block; each wave register-blocks
// FOUR M-tiles so one B fragment feeds 4 WMMAs (4x matrix-op density).
__global__ __launch_bounds__(256) void k_fc(const _Float16* __restrict__ cum1h,
                                            const _Float16* __restrict__ h_allh,
                                            const _Float16* __restrict__ fcwh,
                                            const float* __restrict__ fc_b,
                                            float* __restrict__ res) {
    __shared__ _Float16 lds_b[16 * 512];
    __shared__ float lds_bias[16];
    int nbase = blockIdx.y * 16;
    for (int i = threadIdx.x; i < 16 * 512; i += 256) {
        int r = i >> 9, k = i & 511;
        lds_b[i] = fcwh[((size_t)(nbase + r)) * 512 + k];
    }
    if (threadIdx.x < 16) lds_bias[threadIdx.x] = fc_b[nbase + threadIdx.x];
    __syncthreads();

    int wave = threadIdx.x >> 5;
    int lane = threadIdx.x & 31;
    int ln = lane & 15, half = lane >> 4;
    int mtile0 = (blockIdx.x * 8 + wave) * 4;          // 4 consecutive M-tiles per wave

    const _Float16* a1p[4];   // cum1 half of A (K 0..255)
    const _Float16* a2p[4];   // h half of A    (K 256..511)
    #pragma unroll
    for (int q = 0; q < 4; ++q) {
        int mrow = (mtile0 + q) * 16 + ln;
        a1p[q] = cum1h  + (size_t)mrow * H_;
        a2p[q] = h_allh + (size_t)mrow * H_;
    }

    v8f c0 = {}, c1 = {}, c2 = {}, c3 = {};
    for (int ks = 0; ks < 16; ++ks) {
        int kb = ks * 32;
        v16h bb;
        #pragma unroll
        for (int e = 0; e < 16; ++e)
            bb[e] = lds_b[ln * 512 + kb + (e & 7) + 8 * half + 16 * (e >> 3)];

        v16h a0, a1, a2, a3;
        #pragma unroll
        for (int e = 0; e < 16; ++e) {
            int koff = (e & 7) + 8 * half + 16 * (e >> 3);
            if (ks < 8) {
                a0[e] = a1p[0][kb + koff];
                a1[e] = a1p[1][kb + koff];
                a2[e] = a1p[2][kb + koff];
                a3[e] = a1p[3][kb + koff];
            } else {
                a0[e] = a2p[0][kb - 256 + koff];
                a1[e] = a2p[1][kb - 256 + koff];
                a2[e] = a2p[2][kb - 256 + koff];
                a3[e] = a2p[3][kb - 256 + koff];
            }
        }
        c0 = __builtin_amdgcn_wmma_f32_16x16x32_f16(false, a0, false, bb, (short)0, c0, false, false);
        c1 = __builtin_amdgcn_wmma_f32_16x16x32_f16(false, a1, false, bb, (short)0, c1, false, false);
        c2 = __builtin_amdgcn_wmma_f32_16x16x32_f16(false, a2, false, bb, (short)0, c2, false, false);
        c3 = __builtin_amdgcn_wmma_f32_16x16x32_f16(false, a3, false, bb, (short)0, c3, false, false);
    }
    float bias = lds_bias[ln];
    int col = nbase + ln;
    #pragma unroll
    for (int q = 0; q < 4; ++q) {
        v8f cc = (q == 0) ? c0 : (q == 1) ? c1 : (q == 2) ? c2 : c3;
        #pragma unroll
        for (int v = 0; v < 8; ++v) {
            int row = (mtile0 + q) * 16 + v + 8 * half;
            res[(size_t)row * NUMC + col] = sigmoidf_(cc[v] + bias);
        }
    }
}

// ---------------------------------------------------------------- launch
extern "C" void kernel_launch(void* const* d_in, const int* in_sizes, int n_in,
                              void* d_out, int out_size, void* d_ws, size_t ws_size,
                              hipStream_t stream) {
    const int*   skill      = (const int*)  d_in[0];
    const int*   answer     = (const int*)  d_in[1];
    const float* skill_emb  = (const float*)d_in[2];
    const float* answer_emb = (const float*)d_in[3];
    const float* Wih        = (const float*)d_in[4];
    const float* Whh        = (const float*)d_in[5];
    const float* bih        = (const float*)d_in[6];
    const float* bhh        = (const float*)d_in[7];
    const float* mlp_w      = (const float*)d_in[8];
    const float* mlp_b      = (const float*)d_in[9];
    const float* sim_w      = (const float*)d_in[10];
    const float* fc_w       = (const float*)d_in[11];
    const float* fc_b       = (const float*)d_in[12];

    const size_t RES_ELEMS = (size_t)M_ * NUMC;
    float* res_out = (float*)d_out;
    float* sae_out = (float*)d_out + RES_ELEMS;

    char* w = (char*)d_ws;
    size_t off = 0;
    auto alloc = [&](size_t bytes) -> void* {
        void* p = w + off;
        off = (off + bytes + 255) & ~(size_t)255;
        return p;
    };
    _Float16* saeh    = (_Float16*)alloc((size_t)M_ * D_ * 2);
    _Float16* Wihh    = (_Float16*)alloc((size_t)G4H * D_ * 2);
    _Float16* Whhh    = (_Float16*)alloc((size_t)G4H * H_ * 2);
    _Float16* fcwh    = (_Float16*)alloc((size_t)NUMC * (2 * H_) * 2);
    float*    biasSum = (float*)   alloc((size_t)G4H * 4);
    float*    xg      = (float*)   alloc((size_t)M_ * G4H * 4);
    float*    h_all   = (float*)   alloc((size_t)M_ * H_ * 4);
    _Float16* h_allh  = (_Float16*)alloc((size_t)M_ * H_ * 2);
    _Float16* cum1h   = (_Float16*)alloc((size_t)M_ * H_ * 2);
    float*    scores  = (float*)   alloc((size_t)M_ * 4);

    // weight conversions
    k_cvt_f16<<<512, 256, 0, stream>>>(Wih,  Wihh, G4H * D_);
    k_cvt_f16<<<512, 256, 0, stream>>>(Whh,  Whhh, G4H * H_);
    k_cvt_f16<<<1024, 256, 0, stream>>>(fc_w, fcwh, NUMC * 2 * H_);
    k_bias_sum<<<4, 256, 0, stream>>>(bih, bhh, biasSum);

    // embedding + concat (also writes sae f32 output)
    k_embed<<<M_, 128, 0, stream>>>(skill, answer, skill_emb, answer_emb, sae_out, saeh);

    // xg = sae @ Wih^T + bias   [WMMA]
    k_xg<<<dim3(M_ / 16, G4H / (16 * 8)), 256, 0, stream>>>(saeh, Wihh, biasSum, xg);

    // LSTM recurrence           [WMMA]
    k_lstm<<<B_ / 16, 512, 0, stream>>>(xg, Whhh, h_all, h_allh);

    // attention scores + causal cumulative scan
    k_scores<<<M_, 128, 0, stream>>>(h_all, mlp_w, mlp_b, sim_w, scores);
    k_rowscan<<<B_, 256, 0, stream>>>(scores, h_all, cum1h);

    // final FC + sigmoid        [WMMA] — 4 M-tiles per wave
    k_fc<<<dim3(M_ / (16 * 8 * 4), NUMC / 16), 256, 0, stream>>>(cum1h, h_allh, fcwh, fc_b, res_out);

    (void)in_sizes; (void)n_in; (void)out_size; (void)ws_size;
}
